// CrossAttentionBlock_81896436400601
// MI455X (gfx1250) — compile-verified, matrix operand order checked by reading, not run on hardware
//
#include <hip/hip_runtime.h>
#include <hip/hip_bf16.h>

typedef __attribute__((ext_vector_type(16))) __bf16 v16bf;
typedef __attribute__((ext_vector_type(8)))  float  v8f;

#define D_MODEL  1024
#define N_HEADS  16
#define HEAD_DIM 64
#define SEQ      2048
#define BATCH    4
#define MROWS    (BATCH * SEQ)   // 8192

static __device__ __forceinline__ v8f wmma_bf16(v16bf a, v16bf b, v8f c) {
  return __builtin_amdgcn_wmma_f32_16x16x32_bf16(false, a, false, b, (short)0, c,
                                                 false, false);
}

static __device__ __forceinline__ v8f vzero() {
  v8f z;
#pragma unroll
  for (int i = 0; i < 8; ++i) z[i] = 0.f;
  return z;
}

// ---------------------------------------------------------------------------
// Stage 1: fused LayerNorm -> bf16.  One block (256 threads) per row of 1024.
// ---------------------------------------------------------------------------
__global__ void __launch_bounds__(256)
ln_bf16_kernel(const float* __restrict__ x, const float* __restrict__ w,
               const float* __restrict__ bb, __bf16* __restrict__ out) {
  __shared__ float red[256];
  const int row = blockIdx.x;
  const int t   = threadIdx.x;
  const float4 xv = reinterpret_cast<const float4*>(x + (size_t)row * D_MODEL)[t];
  red[t] = xv.x + xv.y + xv.z + xv.w;
  __syncthreads();
  for (int off = 128; off > 0; off >>= 1) {
    if (t < off) red[t] += red[t + off];
    __syncthreads();
  }
  const float mu = red[0] * (1.f / D_MODEL);
  __syncthreads();
  const float dx = xv.x - mu, dy = xv.y - mu, dz = xv.z - mu, dw = xv.w - mu;
  red[t] = dx * dx + dy * dy + dz * dz + dw * dw;
  __syncthreads();
  for (int off = 128; off > 0; off >>= 1) {
    if (t < off) red[t] += red[t + off];
    __syncthreads();
  }
  const float rstd = rsqrtf(red[0] * (1.f / D_MODEL) + 1e-5f);
  const float4 wv = reinterpret_cast<const float4*>(w)[t];
  const float4 bv = reinterpret_cast<const float4*>(bb)[t];
  union { __bf16 h[4]; uint2 u; } p;
  p.h[0] = (__bf16)(dx * rstd * wv.x + bv.x);
  p.h[1] = (__bf16)(dy * rstd * wv.y + bv.y);
  p.h[2] = (__bf16)(dz * rstd * wv.z + bv.z);
  p.h[3] = (__bf16)(dw * rstd * wv.w + bv.w);
  reinterpret_cast<uint2*>(out + (size_t)row * D_MODEL)[t] = p.u;
}

// ---------------------------------------------------------------------------
// Stage 2: fp32 -> bf16 weight conversion.
// ---------------------------------------------------------------------------
__global__ void __launch_bounds__(256)
f32_to_bf16_kernel(const float* __restrict__ x, __bf16* __restrict__ y, int n4) {
  const int i = blockIdx.x * blockDim.x + threadIdx.x;
  if (i < n4) {
    const float4 v = reinterpret_cast<const float4*>(x)[i];
    union { __bf16 h[4]; uint2 u; } p;
    p.h[0] = (__bf16)v.x; p.h[1] = (__bf16)v.y;
    p.h[2] = (__bf16)v.z; p.h[3] = (__bf16)v.w;
    reinterpret_cast<uint2*>(y)[i] = p.u;
  }
}

// ---------------------------------------------------------------------------
// Stage 3/5: WMMA GEMM  C[m,n] = sum_k A[m,k] * W[n,k] + bias[n]
// Block = 128 threads (4 waves); wave computes 32(M) x 64(N); K = 1024.
// MODE 0: bf16 out, row-major [M, 1024]          (LDS-staged b128 stores)
// MODE 1: bf16 out, V transposed to [B, H, HD, S] (packed b128 stores)
// MODE 2: fp32 out + residual add                 (LDS-staged b128 ld/st)
// ---------------------------------------------------------------------------
template <int MODE>
__global__ void __launch_bounds__(128)
gemm_wmma_kernel(const __bf16* __restrict__ A, const __bf16* __restrict__ W,
                 const float* __restrict__ bias, void* __restrict__ Cout,
                 const float* __restrict__ resid) {
  constexpr int N = D_MODEL, K = D_MODEL;
  const int lane = threadIdx.x & 31;
  const int wave = threadIdx.x >> 5;
  const int bm   = blockIdx.x >> 2;   // MROWS/32 = 256 M-tiles
  const int bn   = blockIdx.x & 3;    // 4 block-columns of 256
  const int m0   = bm * 32;
  const int n0   = bn * 256 + wave * 64;
  const int lm   = lane & 15;
  const int half = lane >> 4;

  v8f acc[2][4];
#pragma unroll
  for (int mh = 0; mh < 2; ++mh)
#pragma unroll
    for (int s = 0; s < 4; ++s) acc[mh][s] = vzero();

  const __bf16* arow0 = A + (size_t)(m0 + lm) * K + half * 16;
  const __bf16* arow1 = A + (size_t)(m0 + 16 + lm) * K + half * 16;
  const __bf16* wrow[4];
#pragma unroll
  for (int s = 0; s < 4; ++s)
    wrow[s] = W + (size_t)(n0 + s * 16 + lm) * K + half * 16;

  for (int k = 0; k < K; k += 32) {
    const v16bf a0 = *reinterpret_cast<const v16bf*>(arow0 + k);
    const v16bf a1 = *reinterpret_cast<const v16bf*>(arow1 + k);
#pragma unroll
    for (int s = 0; s < 4; ++s) {
      const v16bf b = *reinterpret_cast<const v16bf*>(wrow[s] + k);
      acc[0][s] = wmma_bf16(a0, b, acc[0][s]);
      acc[1][s] = wmma_bf16(a1, b, acc[1][s]);
    }
  }

  float bz[4];
#pragma unroll
  for (int s = 0; s < 4; ++s) bz[s] = bias[n0 + s * 16 + lm];

  if constexpr (MODE == 0) {
    __shared__ __attribute__((aligned(16))) __bf16 st[4][32 * 64];
    __bf16* sb = st[wave];
#pragma unroll
    for (int mh = 0; mh < 2; ++mh)
#pragma unroll
      for (int s = 0; s < 4; ++s)
#pragma unroll
        for (int r = 0; r < 8; ++r)
          sb[(mh * 16 + r + 8 * half) * 64 + s * 16 + lm] =
              (__bf16)(acc[mh][s][r] + bz[s]);
    __syncthreads();
    // coalesced copy-out: lane -> one row of 64 bf16 = 8 x b128
    __bf16* C = (__bf16*)Cout;
    const uint4* src = reinterpret_cast<const uint4*>(sb + lane * 64);
    uint4* dst = reinterpret_cast<uint4*>(C + (size_t)(m0 + lane) * N + n0);
#pragma unroll
    for (int i = 0; i < 8; ++i) dst[i] = src[i];
  } else if constexpr (MODE == 1) {
    __bf16* C = (__bf16*)Cout;
    const int bidx = m0 / SEQ;
#pragma unroll
    for (int mh = 0; mh < 2; ++mh) {
      const int ss0 = (m0 % SEQ) + mh * 16 + 8 * half;
#pragma unroll
      for (int s = 0; s < 4; ++s) {
        const int n = n0 + s * 16 + lm;
        const int hh = n >> 6;
        const int dd = n & 63;
        union { __bf16 e[8]; uint4 u; } p;
#pragma unroll
        for (int r = 0; r < 8; ++r) p.e[r] = (__bf16)(acc[mh][s][r] + bz[s]);
        const size_t base =
            (((size_t)bidx * N_HEADS + hh) * HEAD_DIM + dd) * SEQ + ss0;
        *reinterpret_cast<uint4*>(C + base) = p.u;
      }
    }
  } else {
    __shared__ __attribute__((aligned(16))) float stf[4][32 * 64];
    float* sf = stf[wave];
#pragma unroll
    for (int mh = 0; mh < 2; ++mh)
#pragma unroll
      for (int s = 0; s < 4; ++s)
#pragma unroll
        for (int r = 0; r < 8; ++r)
          sf[(mh * 16 + r + 8 * half) * 64 + s * 16 + lm] =
              acc[mh][s][r] + bz[s];
    __syncthreads();
    // coalesced copy-out with residual: lane -> one row of 64 f32 = 16 x b128
    float* C = (float*)Cout;
    const float4* src = reinterpret_cast<const float4*>(sf + lane * 64);
    const size_t base = (size_t)(m0 + lane) * N + n0;
    const float4* rv = reinterpret_cast<const float4*>(resid + base);
    float4* dst = reinterpret_cast<float4*>(C + base);
#pragma unroll
    for (int i = 0; i < 16; ++i) {
      float4 v = src[i];
      const float4 rr = rv[i];
      v.x += rr.x; v.y += rr.y; v.z += rr.z; v.w += rr.w;
      dst[i] = v;
    }
  }
}

// ---------------------------------------------------------------------------
// Stage 4: flash attention.  Block = 128 threads (4 waves); wave handles one
// (b, h, 16 query rows) tile, streaming keys in chunks of 64.
// Softmax runs in row-per-lane layout: scores staged fp32-transposed in LDS,
// each lane owns half a row (32 contiguous floats), so max/sum need only one
// shfl_xor(16) each; P is written back row-major with packed b128 DS stores,
// which directly feeds the WMMA A-fragment reads.
// Q,K row-major [B*S, 1024]; V transposed [B, H, HD, S]; out bf16 row-major.
// ---------------------------------------------------------------------------
__global__ void __launch_bounds__(128)
attn_kernel(const __bf16* __restrict__ Q, const __bf16* __restrict__ Kmat,
            const __bf16* __restrict__ Vt, const unsigned char* __restrict__ mask,
            __bf16* __restrict__ O) {
  __shared__ __attribute__((aligned(16))) float  sS[4][16 * 64];  // 4KB/wave
  __shared__ __attribute__((aligned(32))) __bf16 pB[4][16 * 64];  // 2KB/wave
  const int lane = threadIdx.x & 31;
  const int wave = threadIdx.x >> 5;
  const int wid  = blockIdx.x * 4 + wave;
  const int qt   = wid & (SEQ / 16 - 1);     // 128 q-tiles
  const int h    = (wid >> 7) & (N_HEADS - 1);
  const int b    = wid >> 11;
  const int q0   = qt * 16;
  const int lm   = lane & 15;
  const int half = lane >> 4;

  const __bf16* qp =
      Q + (size_t)(b * SEQ + q0 + lm) * D_MODEL + h * HEAD_DIM + half * 16;
  const v16bf aq0 = *reinterpret_cast<const v16bf*>(qp);
  const v16bf aq1 = *reinterpret_cast<const v16bf*>(qp + 32);

  v8f o[4];
#pragma unroll
  for (int dt = 0; dt < 4; ++dt) o[dt] = vzero();
  float mrow = -1e30f;  // running max of row (lane & 15), replicated in halves
  float lrow = 0.f;     // running denom of row (lane & 15)

  const float scale = 0.125f;  // HEAD_DIM^-0.5
  float* sbufS = sS[wave];
  __bf16* pb   = pB[wave];
  const __bf16* vbase =
      Vt + ((size_t)(b * N_HEADS + h) * HEAD_DIM + lm) * SEQ + half * 16;
  const int rrow = lane & 15;   // row owned by this lane in row-layout
  const int kh2  = lane >> 4;   // which 32-key half of the row

  for (int kk = 0; kk < SEQ; kk += 64) {
    // ---- scores (C-fragment layout), scale + mask fused ----
    float sv[4][8];
#pragma unroll
    for (int t = 0; t < 4; ++t) {
      const __bf16* kp = Kmat + (size_t)(b * SEQ + kk + t * 16 + lm) * D_MODEL +
                         h * HEAD_DIM + half * 16;
      __builtin_prefetch(kp + (size_t)64 * D_MODEL, 0, 3);  // next chunk
      v8f c = vzero();
      c = wmma_bf16(aq0, *reinterpret_cast<const v16bf*>(kp), c);
      c = wmma_bf16(aq1, *reinterpret_cast<const v16bf*>(kp + 32), c);
      const float msk = mask[b * SEQ + kk + t * 16 + lm] ? 0.f : -1e30f;
#pragma unroll
      for (int r = 0; r < 8; ++r) sv[t][r] = c[r] * scale + msk;
    }
    // ---- stage fp32 scores transposed to row-major [row][key] ----
#pragma unroll
    for (int t = 0; t < 4; ++t)
#pragma unroll
      for (int r = 0; r < 8; ++r)
        sbufS[(r + 8 * half) * 64 + t * 16 + lm] = sv[t][r];
    __syncthreads();

    // ---- row-per-lane softmax: lane owns 32 contiguous keys of one row ----
    float4 v4[8];
    const float4* srow =
        reinterpret_cast<const float4*>(sbufS + rrow * 64 + kh2 * 32);
#pragma unroll
    for (int i = 0; i < 8; ++i) v4[i] = srow[i];
    float pmax = -1e30f;
#pragma unroll
    for (int i = 0; i < 8; ++i)
      pmax = fmaxf(pmax, fmaxf(fmaxf(v4[i].x, v4[i].y), fmaxf(v4[i].z, v4[i].w)));
    pmax = fmaxf(pmax, __shfl_xor(pmax, 16, 32));  // other half of the row
    const float mn = fmaxf(mrow, pmax);
    const float cf = __expf(mrow - mn);
    mrow = mn;
    float rs = 0.f;
    union { __bf16 e[8]; uint4 u; } pk[4];
#pragma unroll
    for (int i = 0; i < 8; ++i) {
      const float e0 = __expf(v4[i].x - mn);
      const float e1 = __expf(v4[i].y - mn);
      const float e2 = __expf(v4[i].z - mn);
      const float e3 = __expf(v4[i].w - mn);
      rs += (e0 + e1) + (e2 + e3);
      pk[i >> 1].e[(i & 1) * 4 + 0] = (__bf16)e0;
      pk[i >> 1].e[(i & 1) * 4 + 1] = (__bf16)e1;
      pk[i >> 1].e[(i & 1) * 4 + 2] = (__bf16)e2;
      pk[i >> 1].e[(i & 1) * 4 + 3] = (__bf16)e3;
    }
    rs += __shfl_xor(rs, 16, 32);
    lrow = lrow * cf + rs;
    // packed row-major P store: 4 x b128 per lane
    uint4* pdst = reinterpret_cast<uint4*>(pb + rrow * 64 + kh2 * 32);
#pragma unroll
    for (int i = 0; i < 4; ++i) pdst[i] = pk[i].u;

    // ---- broadcast per-row correction into C-fragment layout, rescale O ----
#pragma unroll
    for (int r = 0; r < 8; ++r) {
      const float cfc = __shfl(cf, r + 8 * half, 32);
#pragma unroll
      for (int dt = 0; dt < 4; ++dt) o[dt][r] *= cfc;
    }
    __syncthreads();

    // ---- A-fragment reads of P, then PV ----
    const v16bf pa0 = *reinterpret_cast<const v16bf*>(pb + lm * 64 + half * 16);
    const v16bf pa1 =
        *reinterpret_cast<const v16bf*>(pb + lm * 64 + 32 + half * 16);
#pragma unroll
    for (int dt = 0; dt < 4; ++dt) {
      const __bf16* vp = vbase + (size_t)dt * 16 * SEQ + kk;
      __builtin_prefetch(vp + 64, 0, 3);  // next chunk
      o[dt] = wmma_bf16(pa0, *reinterpret_cast<const v16bf*>(vp), o[dt]);
      o[dt] = wmma_bf16(pa1, *reinterpret_cast<const v16bf*>(vp + 32), o[dt]);
    }
  }

  // ---- epilogue: normalize, stage tile in LDS, coalesced b128 copy-out ----
  float lrc[8];
#pragma unroll
  for (int r = 0; r < 8; ++r) lrc[r] = __shfl(lrow, r + 8 * half, 32);
  __syncthreads();
#pragma unroll
  for (int dt = 0; dt < 4; ++dt)
#pragma unroll
    for (int r = 0; r < 8; ++r)
      pb[(r + 8 * half) * 64 + dt * 16 + lm] = (__bf16)(o[dt][r] / lrc[r]);
  __syncthreads();
  {
    const int row  = lane >> 1;       // 16 rows, 2 lanes per row
    const int colh = lane & 1;        // 32 bf16 halves
    const uint4* src =
        reinterpret_cast<const uint4*>(pb + row * 64 + colh * 32);
    uint4* dst = reinterpret_cast<uint4*>(
        O + (size_t)(b * SEQ + q0 + row) * D_MODEL + h * HEAD_DIM + colh * 32);
#pragma unroll
    for (int i = 0; i < 4; ++i) dst[i] = src[i];
  }
}

// ---------------------------------------------------------------------------
// Host launch
// ---------------------------------------------------------------------------
extern "C" void kernel_launch(void* const* d_in, const int* in_sizes, int n_in,
                              void* d_out, int out_size, void* d_ws,
                              size_t ws_size, hipStream_t stream) {
  (void)in_sizes; (void)n_in; (void)out_size; (void)ws_size;
  const float* q_in   = (const float*)d_in[0];
  const float* kv_in  = (const float*)d_in[1];
  const unsigned char* mask = (const unsigned char*)d_in[2];
  const float* ln_q_w  = (const float*)d_in[3];
  const float* ln_q_b  = (const float*)d_in[4];
  const float* ln_kv_w = (const float*)d_in[5];
  const float* ln_kv_b = (const float*)d_in[6];
  const float* wq = (const float*)d_in[7];
  const float* bq = (const float*)d_in[8];
  const float* wk = (const float*)d_in[9];
  const float* bk = (const float*)d_in[10];
  const float* wv = (const float*)d_in[11];
  const float* bv = (const float*)d_in[12];
  const float* wo = (const float*)d_in[13];
  const float* bo = (const float*)d_in[14];

  char* ws = (char*)d_ws;
  size_t off = 0;
  auto alloc = [&](size_t bytes) {
    char* p = ws + off;
    off += (bytes + 255) & ~(size_t)255;
    return p;
  };
  const size_t act_b = (size_t)MROWS * D_MODEL * sizeof(__bf16);   // 16 MiB
  const size_t wgt_b = (size_t)D_MODEL * D_MODEL * sizeof(__bf16); // 2 MiB
  __bf16* qn  = (__bf16*)alloc(act_b);
  __bf16* kvn = (__bf16*)alloc(act_b);
  __bf16* wqb = (__bf16*)alloc(wgt_b);
  __bf16* wkb = (__bf16*)alloc(wgt_b);
  __bf16* wvb = (__bf16*)alloc(wgt_b);
  __bf16* wob = (__bf16*)alloc(wgt_b);
  __bf16* Qb  = (__bf16*)alloc(act_b);
  __bf16* Kb  = (__bf16*)alloc(act_b);
  __bf16* Vtb = (__bf16*)alloc(act_b);
  __bf16* Ab  = (__bf16*)alloc(act_b);

  ln_bf16_kernel<<<MROWS, 256, 0, stream>>>(q_in, ln_q_w, ln_q_b, qn);
  ln_bf16_kernel<<<MROWS, 256, 0, stream>>>(kv_in, ln_kv_w, ln_kv_b, kvn);

  const int n4 = D_MODEL * D_MODEL / 4;
  const int cvtg = (n4 + 255) / 256;
  f32_to_bf16_kernel<<<cvtg, 256, 0, stream>>>(wq, wqb, n4);
  f32_to_bf16_kernel<<<cvtg, 256, 0, stream>>>(wk, wkb, n4);
  f32_to_bf16_kernel<<<cvtg, 256, 0, stream>>>(wv, wvb, n4);
  f32_to_bf16_kernel<<<cvtg, 256, 0, stream>>>(wo, wob, n4);

  const dim3 ggrid((MROWS / 32) * (D_MODEL / 256));  // 1024 blocks
  gemm_wmma_kernel<0><<<ggrid, 128, 0, stream>>>(qn,  wqb, bq, Qb,  nullptr);
  gemm_wmma_kernel<0><<<ggrid, 128, 0, stream>>>(kvn, wkb, bk, Kb,  nullptr);
  gemm_wmma_kernel<1><<<ggrid, 128, 0, stream>>>(kvn, wvb, bv, Vtb, nullptr);

  attn_kernel<<<MROWS * N_HEADS / 16 / 4, 128, 0, stream>>>(Qb, Kb, Vtb, mask, Ab);

  gemm_wmma_kernel<2><<<ggrid, 128, 0, stream>>>(Ab, wob, bo, d_out, q_in);
}
